// PointerBlock_9088150798676
// MI455X (gfx1250) — matrix operand order repaired and checked by reading, not compiled, not run on hardware
//
#include <hip/hip_runtime.h>
#include <hip/hip_bf16.h>

// ---------------------------------------------------------------------------
// Problem constants (fixed by the reference)
// ---------------------------------------------------------------------------
#define BATCH   4
#define NN      4096
#define DD      1024
#define BNROWS  (BATCH * NN)      // 16384 rows
#define DH      (DD / 2)          // 512
#define LN_EPS  1e-5f

typedef __attribute__((ext_vector_type(16))) __bf16 v16bf;
typedef __attribute__((ext_vector_type(8)))  float   v8f;
typedef __attribute__((ext_vector_type(8)))  __bf16 bf16x8;
typedef __attribute__((ext_vector_type(4)))  int     v4i;

#if defined(__HIP_DEVICE_COMPILE__) && \
    __has_builtin(__builtin_amdgcn_global_load_async_to_lds_b128) && \
    __has_builtin(__builtin_amdgcn_s_wait_asynccnt)
#define USE_ASYNC 1
// builtin signature: (v4i addrspace(1)*, v4i addrspace(3)*, imm offset, imm cpol)
#define GASYNC_B128(gptr, lptr)                                      \
  __builtin_amdgcn_global_load_async_to_lds_b128(                    \
      (__attribute__((address_space(1))) v4i*)(v4i*)(gptr),          \
      (__attribute__((address_space(3))) v4i*)(v4i*)(lptr), 0, 0)
#else
#define USE_ASYNC 0
#endif

__device__ __forceinline__ float gelu_exact(float x) {
  return 0.5f * x * (1.0f + erff(x * 0.70710678118654752f));
}
__device__ __forceinline__ float sigmoid_f(float x) {
  return 1.0f / (1.0f + expf(-x));
}
__device__ __forceinline__ float wave_sum(float v) {
  #pragma unroll
  for (int off = 16; off > 0; off >>= 1) v += __shfl_xor(v, off, 32);
  return v;
}
__device__ __forceinline__ int clampi(int v, int lo, int hi) {
  return v < lo ? lo : (v > hi ? hi : v);
}

// ---------------------------------------------------------------------------
// fp32 -> bf16 conversion (grid-stride, row-major preserved)
// ---------------------------------------------------------------------------
__global__ void k_f2bf(const float* __restrict__ in, __bf16* __restrict__ out,
                       size_t n) {
  size_t i = (size_t)blockIdx.x * blockDim.x + threadIdx.x;
  size_t stride = (size_t)gridDim.x * blockDim.x;
  for (; i < n; i += stride) out[i] = (__bf16)in[i];
}

// ---------------------------------------------------------------------------
// fp32 [K][Nn] -> bf16 transposed [Nn][K] (32x32 LDS tile transpose)
// ---------------------------------------------------------------------------
__global__ __launch_bounds__(256) void k_conv_T(const float* __restrict__ W,
                                                __bf16* __restrict__ WT,
                                                int K, int Nn) {
  __shared__ float t[32][33];
  const int bk = blockIdx.x * 32, bn = blockIdx.y * 32;
  const int lx = threadIdx.x & 31, ly = threadIdx.x >> 5;   // 32 x 8
  #pragma unroll
  for (int r = 0; r < 32; r += 8)
    t[ly + r][lx] = W[(size_t)(bk + ly + r) * Nn + bn + lx];
  __syncthreads();
  #pragma unroll
  for (int r = 0; r < 32; r += 8)
    WT[(size_t)(bn + ly + r) * K + bk + lx] = (__bf16)t[lx][ly + r];
}

// ---------------------------------------------------------------------------
// Column sum of enc_w1 [D, DH] -> cs[DH]   (enc(h + c) shift trick)
// ---------------------------------------------------------------------------
__global__ void k_colsum(const float* __restrict__ w1, float* __restrict__ cs) {
  int j = threadIdx.x;                 // 512 threads, 1 block
  float s = 0.0f;
  for (int i = 0; i < DD; ++i) s += w1[(size_t)i * DH + j];
  cs[j] = s;
}

// bo[n] = sum_j rt_b2[j] * wo[j][n]
__global__ void k_bo(const float* __restrict__ b2v, const float* __restrict__ wo,
                     float* __restrict__ bo) {
  int n = blockIdx.x * blockDim.x + threadIdx.x;   // 1024
  float s = 0.0f;
  for (int j = 0; j < DD; ++j) s += b2v[j] * wo[(size_t)j * DD + n];
  bo[n] = s;
}

// ---------------------------------------------------------------------------
// Tiled bf16 WMMA GEMM:  C = act( A @ B + bias ),  B given pre-transposed
//   A split-source row-major bf16: cols [0,ksplit) from A0 (ld=ksplit),
//   cols [ksplit,K) from A1 (ld=K-ksplit).   BT row-major bf16 [Nn][K].
//   Out: Cf fp32 [M][Nn] and/or Cb bf16 [M][Nn] and/or CbT bf16 [Nn][M].
// Block tile 128x128, BK=32, 256 thr = 8 waves (4 M x 2 N), wave = 32x64.
// Double-buffered LDS; async global->LDS DMA when the toolchain has it.
// ---------------------------------------------------------------------------
#define TM  128
#define TNB 128
#define TK  32
#define LDT 40          // 32 + 8 pad (bf16); row stride 80B = 5 x 16B aligned

__global__ __launch_bounds__(256) void k_gemm_bf16(
    const __bf16* __restrict__ A0, const __bf16* __restrict__ A1, int ksplit,
    const __bf16* __restrict__ BT, const float* __restrict__ bias,
    float* __restrict__ Cf, __bf16* __restrict__ Cb, __bf16* __restrict__ CbT,
    int M, int K, int Nn, int act_gelu) {
  __shared__ __bf16 As[2][TM * LDT];
  __shared__ __bf16 Bs[2][TNB * LDT];

  const int tid  = threadIdx.x;
  const int wv   = tid >> 5;
  const int lane = tid & 31;
  const int half = lane >> 4;
  const int l16  = lane & 15;
  const int wvm  = wv & 3;        // 4 M-waves * 32 rows
  const int wvn  = wv >> 2;       // 2 N-waves * 64 cols

  const int m0 = blockIdx.y * TM;
  const int n0 = blockIdx.x * TNB;

  v8f acc[2][4];
  #pragma unroll
  for (int mi = 0; mi < 2; ++mi)
    #pragma unroll
    for (int t = 0; t < 4; ++t)
      #pragma unroll
      for (int j = 0; j < 8; ++j) acc[mi][t][j] = 0.0f;

  const int lda1 = K - ksplit;

  // stage copy: 128 rows x 32 elems each side; 512 16B-chunks; 2/thread/side
  auto issue_stage = [&](int kt, int buf) {
    const int k0 = kt * TK;
    const __bf16* Ablk; int lda, kc;
    if (k0 < ksplit) { Ablk = A0; lda = ksplit; kc = k0; }
    else             { Ablk = A1; lda = lda1;   kc = k0 - ksplit; }
    #pragma unroll
    for (int c = 0; c < 2; ++c) {
      const int chunk = c * 256 + tid;
      const int row = chunk >> 2, off = (chunk & 3) * 8;
      const __bf16* g = Ablk + (size_t)(m0 + row) * lda + kc + off;
      __bf16* l = &As[buf][row * LDT + off];
#if USE_ASYNC
      GASYNC_B128(g, l);
#else
      *(bf16x8*)l = *(const bf16x8*)g;
#endif
    }
    #pragma unroll
    for (int c = 0; c < 2; ++c) {
      const int chunk = c * 256 + tid;
      const int row = chunk >> 2, off = (chunk & 3) * 8;
      const __bf16* g = BT + (size_t)(n0 + row) * K + k0 + off;
      __bf16* l = &Bs[buf][row * LDT + off];
#if USE_ASYNC
      GASYNC_B128(g, l);
#else
      *(bf16x8*)l = *(const bf16x8*)g;
#endif
    }
  };

  const int KT = K / TK;
  issue_stage(0, 0);
  for (int kt = 0; kt < KT; ++kt) {
    const int buf = kt & 1;
    if (kt + 1 < KT) {
      issue_stage(kt + 1, buf ^ 1);
#if USE_ASYNC
      __builtin_amdgcn_s_wait_asynccnt(4);   // our stage-kt copies done (in-order)
#endif
    } else {
#if USE_ASYNC
      __builtin_amdgcn_s_wait_asynccnt(0);
#endif
    }
    __syncthreads();

    // A fragments: lane half h reads K chunks [8h,8h+8) and [16+8h,16+8h+8)
    v16bf af[2];
    #pragma unroll
    for (int mi = 0; mi < 2; ++mi) {
      const __bf16* base = &As[buf][(wvm * 32 + mi * 16 + l16) * LDT + 8 * half];
      bf16x8 lo = *(const bf16x8*)base;
      bf16x8 hi = *(const bf16x8*)(base + 16);
      af[mi] = __builtin_shufflevector(lo, hi, 0, 1, 2, 3, 4, 5, 6, 7,
                                       8, 9, 10, 11, 12, 13, 14, 15);
    }
    // B fragments: lane half h reads K chunks [8h,8h+8) and [16+8h,16+8h+8)
    #pragma unroll
    for (int t = 0; t < 4; ++t) {
      const __bf16* bb = &Bs[buf][(wvn * 64 + t * 16 + l16) * LDT + 8 * half];
      bf16x8 lo = *(const bf16x8*)bb;
      bf16x8 hi = *(const bf16x8*)(bb + 16);
      v16bf bfr = __builtin_shufflevector(lo, hi, 0, 1, 2, 3, 4, 5, 6, 7,
                                          8, 9, 10, 11, 12, 13, 14, 15);
      acc[0][t] = __builtin_amdgcn_wmma_f32_16x16x32_bf16(
          false, af[0], false, bfr, (short)0, acc[0][t], false, false);
      acc[1][t] = __builtin_amdgcn_wmma_f32_16x16x32_bf16(
          false, af[1], false, bfr, (short)0, acc[1][t], false, false);
    }
    __syncthreads();
  }

  // epilogue (C layout: M = j + 8*half, N = l16)
  #pragma unroll
  for (int mi = 0; mi < 2; ++mi) {
    #pragma unroll
    for (int t = 0; t < 4; ++t) {
      const int col = n0 + wvn * 64 + t * 16 + l16;
      const float bz = bias ? bias[col] : 0.0f;
      #pragma unroll
      for (int j = 0; j < 8; ++j) {
        const int row = m0 + wvm * 32 + mi * 16 + j + 8 * half;
        float v = acc[mi][t][j] + bz;
        if (act_gelu) v = gelu_exact(v);
        if (Cf)  Cf[(size_t)row * Nn + col] = v;
        if (Cb)  Cb[(size_t)row * Nn + col] = (__bf16)v;
        if (CbT) CbT[(size_t)col * M + row] = (__bf16)v;
      }
    }
  }
}

// ---------------------------------------------------------------------------
// enc second stage for c in {0, 0.1, 0.2}:  e_c = GELU(S + c*cs) . w2 + b2
// ---------------------------------------------------------------------------
__global__ __launch_bounds__(256) void k_branch_logits(
    const float* __restrict__ S, const float* __restrict__ cs,
    const float* __restrict__ w2, const float* __restrict__ b2,
    float* __restrict__ e0, float* __restrict__ e1, float* __restrict__ e2) {
  const int wv = threadIdx.x >> 5, lane = threadIdx.x & 31;
  const int row = blockIdx.x * 8 + wv;
  const float* s = S + (size_t)row * DH;
  float a0 = 0.f, a1 = 0.f, a2 = 0.f;
  #pragma unroll 4
  for (int t = 0; t < DH / 32; ++t) {
    const int j = lane + t * 32;
    const float sv = s[j], c = cs[j], w = w2[j];
    a0 += gelu_exact(sv) * w;
    a1 += gelu_exact(sv + 0.1f * c) * w;
    a2 += gelu_exact(sv + 0.2f * c) * w;
  }
  a0 = wave_sum(a0); a1 = wave_sum(a1); a2 = wave_sum(a2);
  if (lane == 0) {
    const float bb = b2[0];
    e0[row] = a0 + bb; e1[row] = a1 + bb; e2[row] = a2 + bb;
  }
}

// ---------------------------------------------------------------------------
// base strength, num_branches, first hops.  ptrs layout: [(chain*3+hop)][BNROWS]
// ---------------------------------------------------------------------------
__global__ void k_pointer_init(const float* __restrict__ e0,
                               const float* __restrict__ e1,
                               const float* __restrict__ e2,
                               const float* __restrict__ thr,
                               int* __restrict__ ptrs,
                               float* __restrict__ strength) {
  const int i = blockIdx.x * blockDim.x + threadIdx.x;
  if (i >= BNROWS) return;
  const float v0 = e0[i];
  const float bs = sigmoid_f(v0);
  strength[i] = bs;
  const int nb = clampi((int)rintf(bs / thr[0]), 1, 3);
  const float lg[3] = { v0, e1[i], e2[i] };
  #pragma unroll
  for (int b = 0; b < 3; ++b) {
    const int tgt = (int)rintf(sigmoid_f(lg[b]) * (float)(NN - 1));
    ptrs[(size_t)(b * 3) * BNROWS + i] = (b < nb) ? tgt : 0;
  }
}

// ---------------------------------------------------------------------------
// One pointer-chain hop for all 3 chains: gather h row, LayerNorm, dot hop_w
// ---------------------------------------------------------------------------
__global__ __launch_bounds__(256) void k_hop(
    int* __restrict__ ptrs, int hop, const float* __restrict__ h,
    const float* __restrict__ gamma, const float* __restrict__ beta,
    const float* __restrict__ w, const float* __restrict__ hb) {
  const int wv = threadIdx.x >> 5, lane = threadIdx.x & 31;
  const int gr = blockIdx.x * 8 + wv;          // 0 .. 3*BNROWS-1
  const int chain = gr / BNROWS;
  const int r = gr - chain * BNROWS;
  const int batch = r / NN;
  int p = ptrs[(size_t)(chain * 3 + hop - 1) * BNROWS + r];
  p = clampi(p, 0, NN - 1);
  const float* x = h + ((size_t)batch * NN + p) * DD;

  float xv[DD / 32];
  float s = 0.f, s2 = 0.f;
  #pragma unroll 8
  for (int t = 0; t < DD / 32; ++t) {
    const float v = x[lane + t * 32];
    xv[t] = v; s += v; s2 += v * v;
  }
  s = wave_sum(s); s2 = wave_sum(s2);
  const float mu = s * (1.0f / DD);
  const float var = s2 * (1.0f / DD) - mu * mu;
  const float rs = rsqrtf(var + LN_EPS);
  float d = 0.f;
  #pragma unroll 8
  for (int t = 0; t < DD / 32; ++t) {
    const int j = lane + t * 32;
    d += ((xv[t] - mu) * rs * gamma[j] + beta[j]) * w[j];
  }
  d = wave_sum(d);
  if (lane == 0) {
    const float sg = sigmoid_f(d + hb[0]);
    ptrs[(size_t)(chain * 3 + hop) * BNROWS + r] =
        (int)rintf(sg * (float)(NN - 1));
  }
}

// main_ptr = chain0 hop2, written as raw int bits into the fp32 output buffer
__global__ void k_copy_main(const int* __restrict__ ptrs, int* __restrict__ out) {
  const int i = blockIdx.x * blockDim.x + threadIdx.x;
  if (i < BNROWS) out[i] = ptrs[(size_t)2 * BNROWS + i];
}

// ---------------------------------------------------------------------------
// tgt_mean = (1/9) sum over 9 pointer sets of gather(hv).  Block per row.
// ---------------------------------------------------------------------------
__global__ __launch_bounds__(256) void k_aggregate(
    const __bf16* __restrict__ hv, const int* __restrict__ ptrs,
    __bf16* __restrict__ tmean) {
  const int r = blockIdx.x;             // BNROWS blocks
  const int batch = r / NN;
  float acc[4] = {0.f, 0.f, 0.f, 0.f};
  for (int k = 0; k < 9; ++k) {
    int p = clampi(ptrs[(size_t)k * BNROWS + r], 0, NN - 1);
    const __bf16* src = hv + ((size_t)batch * NN + p) * DD;
    #pragma unroll
    for (int t = 0; t < 4; ++t) acc[t] += (float)src[threadIdx.x + t * 256];
  }
  const float inv9 = 1.0f / 9.0f;
  #pragma unroll
  for (int t = 0; t < 4; ++t)
    tmean[(size_t)r * DD + threadIdx.x + t * 256] = (__bf16)(acc[t] * inv9);
}

// ---------------------------------------------------------------------------
// Host orchestration
// ---------------------------------------------------------------------------
static inline char* carve(char*& p, size_t bytes) {
  char* r = p;
  p += (bytes + 255) & ~(size_t)255;
  return r;
}

extern "C" void kernel_launch(void* const* d_in, const int* in_sizes, int n_in,
                              void* d_out, int out_size, void* d_ws, size_t ws_size,
                              hipStream_t stream) {
  (void)in_sizes; (void)n_in; (void)out_size;
  const float* h      = (const float*)d_in[0];
  const float* enc_w1 = (const float*)d_in[1];
  const float* enc_b1 = (const float*)d_in[2];
  const float* enc_w2 = (const float*)d_in[3];
  const float* enc_b2 = (const float*)d_in[4];
  const float* wv     = (const float*)d_in[5];
  const float* rt_w1  = (const float*)d_in[6];
  const float* rt_b1  = (const float*)d_in[7];
  const float* rt_w2  = (const float*)d_in[8];
  const float* rt_b2  = (const float*)d_in[9];
  const float* wo     = (const float*)d_in[10];
  const float* hop_g  = (const float*)d_in[11];
  const float* hop_be = (const float*)d_in[12];
  const float* hop_w  = (const float*)d_in[13];
  const float* hop_b  = (const float*)d_in[14];
  const float* thr    = (const float*)d_in[15];

  float* out_z        = (float*)d_out;                         // [BN, D]
  int*   out_mainptr  = (int*)(out_z + (size_t)BNROWS * DD);   // [BN]
  float* out_strength = (float*)(out_mainptr + BNROWS);        // [BN]

  // ---- workspace layout ----
  char* p = (char*)d_ws;
  __bf16* hb     = (__bf16*)carve(p, (size_t)BNROWS * DD * 2);   // A: h
  __bf16* w1T    = (__bf16*)carve(p, (size_t)DD * DH * 2);       // B^T [DH][DD]
  __bf16* wvT    = (__bf16*)carve(p, (size_t)DD * DD * 2);       // B^T [DD][DD]
  __bf16* rtw1T  = (__bf16*)carve(p, (size_t)2 * DD * DD * 2);   // B^T [DD][2DD]
  __bf16* rtw2b  = (__bf16*)carve(p, (size_t)DD * DD * 2);       // A: rt_w2
  __bf16* woT    = (__bf16*)carve(p, (size_t)DD * DD * 2);       // B^T [DD][DD]
  __bf16* WfT    = (__bf16*)carve(p, (size_t)DD * DD * 2);       // (rt_w2@wo)^T
  float*  S      = (float*)carve(p, (size_t)BNROWS * DH * 4);    // aliased as Hid
  __bf16* Hidb   = (__bf16*)S;                                   // same 33.5MB
  float*  e0     = (float*)carve(p, (size_t)BNROWS * 4);
  float*  e1     = (float*)carve(p, (size_t)BNROWS * 4);
  float*  e2     = (float*)carve(p, (size_t)BNROWS * 4);
  int*    ptrs   = (int*)carve(p, (size_t)9 * BNROWS * 4);
  __bf16* hvb    = (__bf16*)carve(p, (size_t)BNROWS * DD * 2);
  __bf16* tmeanb = (__bf16*)carve(p, (size_t)BNROWS * DD * 2);
  float*  cs     = (float*)carve(p, (size_t)DH * 4);
  float*  bo     = (float*)carve(p, (size_t)DD * 4);
  if ((size_t)(p - (char*)d_ws) > ws_size) return;   // insufficient scratch

  // ---- conversions: A operands plain, B operands transposed ----
  k_f2bf<<<2048, 256, 0, stream>>>(h,     hb,    (size_t)BNROWS * DD);
  k_f2bf<<<512,  256, 0, stream>>>(rt_w2, rtw2b, (size_t)DD * DD);
  k_conv_T<<<dim3(DD / 32, DH / 32),     256, 0, stream>>>(enc_w1, w1T,   DD,     DH);
  k_conv_T<<<dim3(DD / 32, DD / 32),     256, 0, stream>>>(wv,     wvT,   DD,     DD);
  k_conv_T<<<dim3(2 * DD / 32, DD / 32), 256, 0, stream>>>(rt_w1,  rtw1T, 2 * DD, DD);
  k_conv_T<<<dim3(DD / 32, DD / 32),     256, 0, stream>>>(wo,     woT,   DD,     DD);

  // ---- small precomputes ----
  k_colsum<<<1, DH, 0, stream>>>(enc_w1, cs);
  k_bo<<<DD / 256, 256, 0, stream>>>(rt_b2, wo, bo);

  // ---- WfT = (rt_w2 @ wo)^T (fused tail weight), bf16 transposed out ----
  k_gemm_bf16<<<dim3(DD / TNB, DD / TM), 256, 0, stream>>>(
      rtw2b, nullptr, DD, woT, nullptr, nullptr, nullptr, WfT, DD, DD, DD, 0);

  // ---- S = h @ enc_w1 + b1 (pre-GELU), fp32 out ----
  k_gemm_bf16<<<dim3(DH / TNB, BNROWS / TM), 256, 0, stream>>>(
      hb, nullptr, DD, w1T, enc_b1, S, nullptr, nullptr, BNROWS, DD, DH, 0);

  // ---- hv = h @ wv, bf16 out ----
  k_gemm_bf16<<<dim3(DD / TNB, BNROWS / TM), 256, 0, stream>>>(
      hb, nullptr, DD, wvT, nullptr, nullptr, hvb, nullptr, BNROWS, DD, DD, 0);

  // ---- branch logits (shift trick), pointers, chains ----
  k_branch_logits<<<BNROWS / 8, 256, 0, stream>>>(S, cs, enc_w2, enc_b2, e0, e1, e2);
  k_pointer_init<<<(BNROWS + 255) / 256, 256, 0, stream>>>(e0, e1, e2, thr, ptrs,
                                                           out_strength);
  k_hop<<<3 * BNROWS / 8, 256, 0, stream>>>(ptrs, 1, h, hop_g,      hop_be,
                                            hop_w,      hop_b);
  k_hop<<<3 * BNROWS / 8, 256, 0, stream>>>(ptrs, 2, h, hop_g + DD, hop_be + DD,
                                            hop_w + DD, hop_b + 1);
  k_copy_main<<<BNROWS / 256, 256, 0, stream>>>(ptrs, out_mainptr);

  // ---- tgt_mean ----
  k_aggregate<<<BNROWS, 256, 0, stream>>>(hvb, ptrs, tmeanb);

  // ---- Hid = GELU([hv, tgt_mean] @ rt_w1 + rt_b1), split-A, bf16 out ----
  k_gemm_bf16<<<dim3(DD / TNB, BNROWS / TM), 256, 0, stream>>>(
      hvb, tmeanb, DD, rtw1T, rt_b1, nullptr, Hidb, nullptr, BNROWS, 2 * DD, DD, 1);

  // ---- z = Hid @ Wf + bo, fp32 out straight to d_out ----
  k_gemm_bf16<<<dim3(DD / TNB, BNROWS / TM), 256, 0, stream>>>(
      Hidb, nullptr, DD, WfT, bo, out_z, nullptr, nullptr, BNROWS, DD, DD, 0);
}
